// EquivariantGNN_51307679318539
// MI455X (gfx1250) — matrix operand order repaired and checked by reading, not compiled, not run on hardware
//
#include <hip/hip_runtime.h>
#include <hip/hip_bf16.h>

// ---------------- problem constants (match reference) ----------------
#define N_NODES   500000
#define N_FEAT    125
#define HIDDEN    128
#define OUT_DIM   64
#define N_GRAPHS  4096
#define N_TYPES   100

#define TILE_M    128           // nodes per workgroup (8 waves x 16 rows)
#define LDX       136           // ushort (bf16) row stride, mult of 8 -> 16B-aligned rows
#define LDO       68            // float row stride for layer-3 output tile

// packed weight image: W1^T | W2^T | W3^T, each row-major [N][K] bf16, stride LDX
#define WPACK_USHORTS ((HIDDEN + HIDDEN + OUT_DIM) * LDX)   // 320*136 = 43520
#define WPACK_BYTES   (WPACK_USHORTS * 2)                   // 87040
#define TDM_UNITS     (WPACK_BYTES / 8)                     // 10880 x 8B units
#define NPREP_W       (2 * 128 * 128 + 64 * 128)            // 40960 weight elems
#define NPREP_E       (N_TYPES * 128)                       // 12800 padded emb elems

typedef __attribute__((ext_vector_type(16))) __bf16         v16bf;
typedef __attribute__((ext_vector_type(8)))  float          v8f;
typedef __attribute__((ext_vector_type(8)))  unsigned short us8;
typedef __attribute__((ext_vector_type(4)))  unsigned int   u32x4;
typedef __attribute__((ext_vector_type(8)))  int            i32x8;
typedef __attribute__((ext_vector_type(4)))  int            i32x4;

union BF16x16 { v16bf bf; us8 h[2]; };

// float -> bf16 round-to-nearest-even
__device__ __forceinline__ unsigned short f2bf(float f) {
    unsigned int u = __float_as_uint(f);
    unsigned int r = u + 0x7FFFu + ((u >> 16) & 1u);
    return (unsigned short)(r >> 16);
}

// One 16x16 D tile of X[128 rows] x Wt^T, K=128, bf16 in / f32 accum.
// A (16-bit 16x32): lane lm, half h -> row m0+lm; VGPR0-3: K{h*8..+7},
//   VGPR4-7: K{16+h*8..+7}  -> two ds_load_b128.
// B (16-bit 32x16): lane lm, half h -> col n0+lm; VGPR0-3: K{h*8..+7},
//   VGPR4-7: K{16+h*8..+7} -> two ds_load_b128 from transposed weights.
__device__ __forceinline__ v8f wmma_tile_k128(
    const unsigned short* __restrict__ X,
    const unsigned short* __restrict__ Wt,
    int m0, int n0, int lm, int half, float binit)
{
    v8f c = { binit, binit, binit, binit, binit, binit, binit, binit };
    const int arow = m0 + lm;
    const int bcol = n0 + lm;
#pragma unroll
    for (int k0 = 0; k0 < 128; k0 += 32) {
        BF16x16 A, B;
        A.h[0] = *(const us8*)&X[arow * LDX + k0 + half * 8];
        A.h[1] = *(const us8*)&X[arow * LDX + k0 + 16 + half * 8];
        B.h[0] = *(const us8*)&Wt[bcol * LDX + k0 + half * 8];
        B.h[1] = *(const us8*)&Wt[bcol * LDX + k0 + 16 + half * 8];
        c = __builtin_amdgcn_wmma_f32_16x16x32_bf16(
                /*neg_a=*/false, A.bf, /*neg_b=*/false, B.bf,
                /*c_mod=*/(short)0, c, /*reuse_a=*/false, /*reuse_b=*/false);
    }
    return c;
}

// ---- one-shot: bf16 transposed weight image + padded bf16 embedding image ----
__global__ void prep_kernel(const float* __restrict__ W1,
                            const float* __restrict__ W2,
                            const float* __restrict__ W3,
                            const float* __restrict__ emb,
                            unsigned short* __restrict__ wpack,
                            unsigned short* __restrict__ embPad) {
    int i = blockIdx.x * blockDim.x + threadIdx.x;
    if (i < 128 * 128) {                       // W1^T
        int n = i & 127, k = i >> 7;
        wpack[n * LDX + k] = f2bf(W1[k * 128 + n]);
    } else if (i < 2 * 128 * 128) {            // W2^T
        int j = i - 128 * 128;
        int n = j & 127, k = j >> 7;
        wpack[HIDDEN * LDX + n * LDX + k] = f2bf(W2[k * 128 + n]);
    } else if (i < NPREP_W) {                  // W3^T
        int j = i - 2 * 128 * 128;
        int n = j & 63, k = j >> 6;
        wpack[2 * HIDDEN * LDX + n * LDX + k] = f2bf(W3[k * 64 + n]);
    } else if (i < NPREP_W + NPREP_E) {        // embPad[z][0..127]: {0,0,0, emb row}
        int j = i - NPREP_W;
        int zi = j >> 7, c = j & 127;
        embPad[zi * 128 + c] = (c < 3) ? (unsigned short)0
                                       : f2bf(emb[zi * N_FEAT + (c - 3)]);
    }
}

__global__ __launch_bounds__(256) void gnn_mlp_pool_kernel(
    const float* __restrict__ pos,  const int* __restrict__ z,
    const int* __restrict__ batch,
    const unsigned short* __restrict__ wpack,
    const unsigned short* __restrict__ embPad,
    const float* __restrict__ b1,   const float* __restrict__ b2,
    const float* __restrict__ b3,
    float* __restrict__ gsum,       float* __restrict__ gcnt)
{
    // ---- LDS (~121 KB, fits 320 KB WGP LDS) ----
    __shared__ __align__(16) unsigned short sXw[TILE_M * LDX];   // node rows, bf16
    __shared__ __align__(16) unsigned short sW [WPACK_USHORTS];  // W1^T|W2^T|W3^T (then f32 X3)
    __shared__ float sB1[HIDDEN], sB2[HIDDEN], sB3[OUT_DIM];
    __shared__ int   sBatch[TILE_M];

    const int t    = threadIdx.x;
    const int lane = t & 31;
    const int wave = t >> 5;
    const int half = lane >> 4;
    const int lm   = lane & 15;
    const int m0   = wave * 16;
    const long tileBase = (long)blockIdx.x * TILE_M;

    // ---- kick off async weight staging (Tensor Data Mover), wave 0 only ----
#if __has_builtin(__builtin_amdgcn_tensor_load_to_lds)
    if (wave == 0) {
        const unsigned long long ga = (unsigned long long)(const void*)wpack;
        const unsigned la = (unsigned)(unsigned long long)(const void*)sW;
        // D# group0: count=1 | lds_addr | global_addr[56:0] | type=2
        u32x4 g0 = { 1u, la, (unsigned)ga,
                     (unsigned)((ga >> 32) & 0x01FFFFFFu) | (2u << 30) };
        // D# group1: data_size=8B, 1-D tile of TDM_UNITS contiguous units
        i32x8 g1 = { (int)(3u << 16),                  // data_size = 3 (8 bytes)
                     (int)((unsigned)TDM_UNITS << 16), // tensor_dim0[15:0] @ 63:48
                     (int)(1u << 16),                  // tensor_dim0 hi=0, tensor_dim1=1
                     (int)((unsigned)TDM_UNITS << 16), // tensor_dim1 hi=0, tile_dim0
                     1,                                // tile_dim1=1, tile_dim2=0
                     (int)TDM_UNITS,                   // tensor_dim0_stride[31:0]
                     0, 0 };
        i32x4 gz = { 0, 0, 0, 0 };
#if __has_include(<hip/amd_detail/amd_gfx1250_TDM.h>)
        i32x8 gz8 = { 0, 0, 0, 0, 0, 0, 0, 0 };
        __builtin_amdgcn_tensor_load_to_lds(g0, g1, gz, gz, gz8, 0);
#else
        __builtin_amdgcn_tensor_load_to_lds(g0, g1, gz, gz, 0);
#endif
    }
#else
    for (int i = t; i < WPACK_USHORTS / 8; i += 256)
        ((us8*)sW)[i] = ((const us8*)wpack)[i];
#endif

    // ---- biases ----
    if (t < 128) { sB1[t] = b1[t]; sB2[t] = b2[t]; }
    if (t < 64)  { sB3[t] = b3[t]; }

    // ---- stage input tile: vector-copy padded bf16 emb row, overlay pos ----
    {
        const int r = t >> 1;            // local row 0..127
        const int h = t & 1;             // row half (64 cols each)
        const long node = tileBase + r;
        const bool valid = (node < (long)N_NODES);
        us8* dst = (us8*)&sXw[r * LDX + h * 64];
        if (valid) {
            const int zi = z[node];
            if (h == 0) sBatch[r] = batch[node];
            const us8* src = (const us8*)&embPad[zi * 128 + h * 64];
#pragma unroll
            for (int j = 0; j < 8; ++j) dst[j] = src[j];
            if (h == 0) {   // same-wave LDS ordering: these land after the b128 store
                sXw[r * LDX + 0] = f2bf(pos[node * 3 + 0]);
                sXw[r * LDX + 1] = f2bf(pos[node * 3 + 1]);
                sXw[r * LDX + 2] = f2bf(pos[node * 3 + 2]);
            }
        } else {
            if (h == 0) sBatch[r] = -1;
            us8 zz = { 0, 0, 0, 0, 0, 0, 0, 0 };
#pragma unroll
            for (int j = 0; j < 8; ++j) dst[j] = zz;
        }
        if (tileBase + TILE_M < (long)N_NODES)
            __builtin_prefetch(pos + (tileBase + TILE_M) * 3 + t, 0, 1);
    }

#if __has_builtin(__builtin_amdgcn_tensor_load_to_lds)
    if (wave == 0) __builtin_amdgcn_s_wait_tensorcnt(0);
#endif
    __syncthreads();

    const unsigned short* sW1t = sW;
    const unsigned short* sW2t = sW + HIDDEN * LDX;
    const unsigned short* sW3t = sW + 2 * HIDDEN * LDX;

    // ==== fused MLP: each wave owns its 16 rows -> no inter-wave barriers ====
    {   // layer 1: relu(X*W1 + b1), bf16 back in place
        v8f acc[8];
#pragma unroll
        for (int nj = 0; nj < 8; ++nj)
            acc[nj] = wmma_tile_k128(sXw, sW1t, m0, nj * 16, lm, half, sB1[nj * 16 + lm]);
#pragma unroll
        for (int nj = 0; nj < 8; ++nj)
#pragma unroll
            for (int r = 0; r < 8; ++r) {
                float v = fmaxf(acc[nj][r], 0.0f);
                sXw[(m0 + half * 8 + r) * LDX + nj * 16 + lm] = f2bf(v);
            }
    }
    {   // layer 2: relu(H*W2 + b2), in place
        v8f acc[8];
#pragma unroll
        for (int nj = 0; nj < 8; ++nj)
            acc[nj] = wmma_tile_k128(sXw, sW2t, m0, nj * 16, lm, half, sB2[nj * 16 + lm]);
#pragma unroll
        for (int nj = 0; nj < 8; ++nj)
#pragma unroll
            for (int r = 0; r < 8; ++r) {
                float v = fmaxf(acc[nj][r], 0.0f);
                sXw[(m0 + half * 8 + r) * LDX + nj * 16 + lm] = f2bf(v);
            }
    }
    // layer 3: H*W3 + b3, f32 kept in registers
    v8f acc3[4];
#pragma unroll
    for (int nj = 0; nj < 4; ++nj)
        acc3[nj] = wmma_tile_k128(sXw, sW3t, m0, nj * 16, lm, half, sB3[nj * 16 + lm]);

    __syncthreads();            // weights dead -> reuse sW as f32 output tile
    {
        float* X3F = (float*)sW;
#pragma unroll
        for (int nj = 0; nj < 4; ++nj)
#pragma unroll
            for (int r = 0; r < 8; ++r)
                X3F[(m0 + half * 8 + r) * LDO + nj * 16 + lm] = acc3[nj][r];
    }
    __syncthreads();

    // ==== sorted-run segmented reduction -> global accumulators ====
    {
        const float* X3F = (const float*)sW;
        const int c = t & 63;
        const int q = t >> 6;
        float accv = 0.0f;
        int cur = -1, runlen = 0;
        for (int n = q * 32; n < q * 32 + 32; ++n) {
            const int b = sBatch[n];
            if (b != cur) {
                if (cur >= 0) {
                    atomicAdd(&gsum[cur * OUT_DIM + c], accv);
                    if (c == 0) atomicAdd(&gcnt[cur], (float)runlen);
                }
                cur = b; accv = 0.0f; runlen = 0;
            }
            if (b >= 0) { accv += X3F[n * LDO + c]; ++runlen; }
        }
        if (cur >= 0) {
            atomicAdd(&gsum[cur * OUT_DIM + c], accv);
            if (c == 0) atomicAdd(&gcnt[cur], (float)runlen);
        }
    }
}

__global__ void zero_ws_kernel(float* __restrict__ ws, int n) {
    int i = blockIdx.x * blockDim.x + threadIdx.x;
    if (i < n) ws[i] = 0.0f;
}

__global__ void finalize_kernel(const float* __restrict__ gsum,
                                const float* __restrict__ gcnt,
                                float* __restrict__ out) {
    int i = blockIdx.x * blockDim.x + threadIdx.x;
    if (i < N_GRAPHS * OUT_DIM) {
        int g = i >> 6;
        out[i] = gsum[i] / fmaxf(gcnt[g], 1.0f);
    }
}

extern "C" void kernel_launch(void* const* d_in, const int* in_sizes, int n_in,
                              void* d_out, int out_size, void* d_ws, size_t ws_size,
                              hipStream_t stream) {
    const float* pos   = (const float*)d_in[0];
    const int*   z     = (const int*)  d_in[1];
    const int*   batch = (const int*)  d_in[2];
    const float* emb   = (const float*)d_in[3];
    const float* W1    = (const float*)d_in[4];
    const float* b1    = (const float*)d_in[5];
    const float* W2    = (const float*)d_in[6];
    const float* b2    = (const float*)d_in[7];
    const float* W3    = (const float*)d_in[8];
    const float* b3    = (const float*)d_in[9];

    float* gsum = (float*)d_ws;                              // [N_GRAPHS*OUT_DIM]
    float* gcnt = gsum + N_GRAPHS * OUT_DIM;                 // [N_GRAPHS]
    unsigned short* wpack  = (unsigned short*)(gcnt + N_GRAPHS);   // bf16 weight image
    unsigned short* embPad = wpack + WPACK_USHORTS;                // bf16 padded emb

    const int nzero = N_GRAPHS * OUT_DIM + N_GRAPHS;
    zero_ws_kernel<<<(nzero + 255) / 256, 256, 0, stream>>>(gsum, nzero);

    const int nprep = NPREP_W + NPREP_E;                     // 53760
    prep_kernel<<<(nprep + 255) / 256, 256, 0, stream>>>(W1, W2, W3, emb,
                                                         wpack, embPad);

    const int tiles = (N_NODES + TILE_M - 1) / TILE_M;       // 3907
    gnn_mlp_pool_kernel<<<tiles, 256, 0, stream>>>(
        pos, z, batch, wpack, embPad, b1, b2, b3, gsum, gcnt);

    finalize_kernel<<<(N_GRAPHS * OUT_DIM + 255) / 256, 256, 0, stream>>>(
        gsum, gcnt, (float*)d_out);
}
